// AddNet_54065048322938
// MI455X (gfx1250) — matrix operand order repaired and checked by reading, not compiled, not run on hardware
//
#include <hip/hip_runtime.h>
#include <hip/hip_bf16.h>
#include <math.h>

// ---------------------------------------------------------------------------
// T=1024, B=64, INP=512, HS=1024, OUT=512 recurrent net, factorized:
//   GEMM1  a1/c1 = f(X @ Wi1^T)          [65536x512x1024]
//   GEMM2  p1    = X @ W1^T + b1         [65536x512x1024]
//   GEMM3  c2    = f(a1 @ Wi2^T)         [65536x1024x1024]
//   scan1  h1[t] = h1[t-1]*c1[t]+p1[t];  an1=tanh(h1)  (elementwise, in place)
//   GEMM4  u2    = an1 @ W2^T + b2       [65536x1024x1024]
//   scan2  h2[t] = h2[t-1]*c2[t]+u2[t];  H2T=tanh(h2[T-1])
//   GEMM5  out   = H2T @ W3^T + b3       [64x1024x512] f32
// ~412 GFLOP f16 WMMA vs ~1.5-2 GB HBM traffic @23.3TB/s -> balanced; big
// GEMMs use async-to-LDS double buffering to hide global latency.
// ---------------------------------------------------------------------------

typedef _Float16 half_t;
typedef __attribute__((ext_vector_type(16))) _Float16     v16h;
typedef __attribute__((ext_vector_type(8)))  float        v8f;
typedef __attribute__((ext_vector_type(4)))  unsigned int v4u;

#define T_STEPS 1024
#define BATCH   64
#define INPD    512
#define HSD     1024
#define OUTD    512
#define MROWS   (T_STEPS * BATCH)   // 65536

#define KT     64                   // K per LDS stage (2 WMMA slices of 32)
#define LDSR2  72                   // LDS row stride in halfs (64 + 8 pad)
#define STAGEH (128 * LDSR2)        // halfs per matrix per stage

// epilogue kinds
#define EPI_SIG_GATE 0
#define EPI_GATE     1
#define EPI_BIAS_F16 2

struct H2x { v4u lo, hi; };

static __device__ __forceinline__ float fast_sigmoid(float y) {
    return 1.0f / (1.0f + __expf(-y));
}
static __device__ __forceinline__ float fast_tanh(float x) {
    // 2*sigmoid(2x)-1: exact +/-1 limits, no NaN from overflow
    return 2.0f / (1.0f + __expf(-2.0f * x)) - 1.0f;
}

__global__ void cvt_f32_f16(const float* __restrict__ in, half_t* __restrict__ out, int n) {
    int i = blockIdx.x * blockDim.x + threadIdx.x;
    int stride = gridDim.x * blockDim.x;
    for (; i < n; i += stride) out[i] = (half_t)in[i];
}

static __device__ __forceinline__ void async_ld_b128(unsigned lds_off, const half_t* g) {
    // per-lane 16B global -> LDS, tracked by ASYNCcnt
    asm volatile("global_load_async_to_lds_b128 %0, %1, off"
                 :: "v"(lds_off), "v"(g) : "memory");
}

// ---------------------------------------------------------------------------
// Async double-buffered WMMA GEMM: Y[m,n] = sum_k A[m,k]*W[n,k]
// A: MxK f16 row-major, W: NxK f16 row-major. M,N multiples of 128; K of 64.
// 256 threads = 8 waves (4 in M x 2 in N); wave tile 32x64 = 2x4 WMMA accs.
// ---------------------------------------------------------------------------
template <int EPI>
__global__ __launch_bounds__(256) void gemm_wmma_async(
    const half_t* __restrict__ A,
    const half_t* __restrict__ W,
    int M, int N, int K,
    const float* __restrict__ bias,
    const float* __restrict__ thr,
    const float* __restrict__ dec,
    half_t* __restrict__ out1,
    half_t* __restrict__ out2)
{
    __shared__ __align__(16) half_t As[2][STAGEH];
    __shared__ __align__(16) half_t Bs[2][STAGEH];

    const int tid   = threadIdx.x;
    const int lane  = tid & 31;
    const int wave  = tid >> 5;
    const int waveM = wave & 3;
    const int waveN = wave >> 2;
    const int lm    = lane & 15;
    const int hi    = lane >> 4;

    const int rowBase = blockIdx.y * 128;
    const int colBase = blockIdx.x * 128;

    const unsigned ldsA[2] = { (unsigned)(size_t)&As[0][0], (unsigned)(size_t)&As[1][0] };
    const unsigned ldsB[2] = { (unsigned)(size_t)&Bs[0][0], (unsigned)(size_t)&Bs[1][0] };

    // copy mapping: 4 chunks (16B) per thread per matrix per stage
    const int crow = tid >> 1;                // combined below per i
    (void)crow;

    v8f acc[2][4];
    {
        v8f z = {};
        #pragma unroll
        for (int i = 0; i < 2; ++i)
            #pragma unroll
            for (int j = 0; j < 4; ++j) acc[i][j] = z;
    }

    int aOff[2], bOff[4];
    #pragma unroll
    for (int i = 0; i < 2; ++i)
        aOff[i] = (waveM * 32 + i * 16 + lm) * LDSR2 + hi * 8;
    #pragma unroll
    for (int j = 0; j < 4; ++j)
        bOff[j] = (waveN * 64 + j * 16 + lm) * LDSR2 + hi * 8;

    const int kTiles = K / KT;

    auto issue_stage = [&](int kt, int buf) {
        const int k0 = kt * KT;
        #pragma unroll
        for (int i = 0; i < 4; ++i) {
            const int lin = tid + i * 256;        // 0..1023
            const int row = lin >> 3;             // 0..127
            const int ch  = (lin & 7) * 8;        // half offset in row
            const unsigned lOff = (unsigned)((row * LDSR2 + ch) * 2);
            async_ld_b128(ldsA[buf] + lOff, A + (size_t)(rowBase + row) * K + k0 + ch);
            async_ld_b128(ldsB[buf] + lOff, W + (size_t)(colBase + row) * K + k0 + ch);
        }
    };

    issue_stage(0, 0);

    for (int kt = 0; kt < kTiles; ++kt) {
        if (kt + 1 < kTiles) {
            issue_stage(kt + 1, (kt + 1) & 1);
            asm volatile("s_wait_asynccnt 0x8" ::: "memory");
        } else {
            asm volatile("s_wait_asynccnt 0x0" ::: "memory");
        }
        __syncthreads();   // whole stage kt resident in LDS for all waves

        const half_t* Ab = &As[kt & 1][0];
        const half_t* Bb = &Bs[kt & 1][0];

        #pragma unroll
        for (int s = 0; s < 2; ++s) {
            const int sc = s * 32;
            v16h afrag[2], bfrag[4];
            #pragma unroll
            for (int i = 0; i < 2; ++i) {
                H2x t;
                t.lo = *(const v4u*)(Ab + aOff[i] + sc);
                t.hi = *(const v4u*)(Ab + aOff[i] + sc + 16);
                afrag[i] = __builtin_bit_cast(v16h, t);
            }
            #pragma unroll
            for (int j = 0; j < 4; ++j) {
                H2x t;
                t.lo = *(const v4u*)(Bb + bOff[j] + sc);
                t.hi = *(const v4u*)(Bb + bOff[j] + sc + 16);
                bfrag[j] = __builtin_bit_cast(v16h, t);
            }
            #pragma unroll
            for (int i = 0; i < 2; ++i)
                #pragma unroll
                for (int j = 0; j < 4; ++j)
                    acc[i][j] = __builtin_amdgcn_wmma_f32_16x16x32_f16(
                        false, afrag[i], false, bfrag[j],
                        (short)0, acc[i][j], false, false);
        }
        __syncthreads();   // reads of buf (kt&1) done before it is refilled
    }

    // epilogue: D lane l -> col=l%16, row=r+(l>=16?8:0)
    #pragma unroll
    for (int i = 0; i < 2; ++i) {
        #pragma unroll
        for (int j = 0; j < 4; ++j) {
            const int col = colBase + waveN * 64 + j * 16 + lm;
            const float bn = bias[col];
            float tn = 0.0f, dn = 0.0f;
            if (EPI == EPI_SIG_GATE || EPI == EPI_GATE) { tn = thr[col]; dn = dec[col]; }
            #pragma unroll
            for (int r = 0; r < 8; ++r) {
                const int row = rowBase + waveM * 32 + i * 16 + hi * 8 + r;
                const size_t idx = (size_t)row * N + col;
                const float y = acc[i][j][r] + bn;
                if (EPI == EPI_SIG_GATE) {
                    const float a = fast_sigmoid(y);
                    out1[idx] = (half_t)a;
                    out2[idx] = (half_t)((a > tn) ? dn : 0.0f);
                } else if (EPI == EPI_GATE) {
                    const float a = fast_sigmoid(y);
                    out2[idx] = (half_t)((a > tn) ? dn : 0.0f);
                } else {
                    out1[idx] = (half_t)y;
                }
            }
        }
    }
}

// ---------------------------------------------------------------------------
// Small guarded synchronous GEMM (readout): outf = A @ W^T + bias, f32 out.
// M may be < 128; N multiple of 128; K multiple of 32.
// ---------------------------------------------------------------------------
#define LDSR 40
__global__ __launch_bounds__(256) void gemm_wmma_small(
    const half_t* __restrict__ A,
    const half_t* __restrict__ W,
    int M, int N, int K,
    const float* __restrict__ bias,
    float* __restrict__ outf)
{
    __shared__ __align__(16) half_t As[128 * LDSR];
    __shared__ __align__(16) half_t Bs[128 * LDSR];

    const int tid   = threadIdx.x;
    const int lane  = tid & 31;
    const int wave  = tid >> 5;
    const int waveM = wave & 3;
    const int waveN = wave >> 2;
    const int lm    = lane & 15;
    const int hi    = lane >> 4;

    const int rowBase = blockIdx.y * 128;
    const int colBase = blockIdx.x * 128;

    const int r0 = tid >> 2;
    const int c0 = (tid & 3) * 8;

    v8f acc[2][4];
    {
        v8f z = {};
        #pragma unroll
        for (int i = 0; i < 2; ++i)
            #pragma unroll
            for (int j = 0; j < 4; ++j) acc[i][j] = z;
    }

    int aOff[2], bOff[4];
    #pragma unroll
    for (int i = 0; i < 2; ++i) aOff[i] = (waveM * 32 + i * 16 + lm) * LDSR + hi * 8;
    #pragma unroll
    for (int j = 0; j < 4; ++j) bOff[j] = (waveN * 64 + j * 16 + lm) * LDSR + hi * 8;

    const int kTiles = K >> 5;
    for (int kt = 0; kt < kTiles; ++kt) {
        const int k0 = kt << 5;
        v4u z = {};
        const int ra = rowBase + r0;
        const int rb = ra + 64;
        v4u a0 = (ra < M) ? *(const v4u*)(A + (size_t)ra * K + k0 + c0) : z;
        v4u a1 = (rb < M) ? *(const v4u*)(A + (size_t)rb * K + k0 + c0) : z;
        v4u b0 = *(const v4u*)(W + (size_t)(colBase + r0)      * K + k0 + c0);
        v4u b1 = *(const v4u*)(W + (size_t)(colBase + r0 + 64) * K + k0 + c0);
        __syncthreads();
        *(v4u*)(As + r0 * LDSR + c0)        = a0;
        *(v4u*)(As + (r0 + 64) * LDSR + c0) = a1;
        *(v4u*)(Bs + r0 * LDSR + c0)        = b0;
        *(v4u*)(Bs + (r0 + 64) * LDSR + c0) = b1;
        __syncthreads();

        v16h afrag[2], bfrag[4];
        #pragma unroll
        for (int i = 0; i < 2; ++i) {
            H2x t;
            t.lo = *(const v4u*)(As + aOff[i]);
            t.hi = *(const v4u*)(As + aOff[i] + 16);
            afrag[i] = __builtin_bit_cast(v16h, t);
        }
        #pragma unroll
        for (int j = 0; j < 4; ++j) {
            H2x t;
            t.lo = *(const v4u*)(Bs + bOff[j]);
            t.hi = *(const v4u*)(Bs + bOff[j] + 16);
            bfrag[j] = __builtin_bit_cast(v16h, t);
        }
        #pragma unroll
        for (int i = 0; i < 2; ++i)
            #pragma unroll
            for (int j = 0; j < 4; ++j)
                acc[i][j] = __builtin_amdgcn_wmma_f32_16x16x32_f16(
                    false, afrag[i], false, bfrag[j],
                    (short)0, acc[i][j], false, false);
    }

    #pragma unroll
    for (int i = 0; i < 2; ++i) {
        #pragma unroll
        for (int j = 0; j < 4; ++j) {
            const int col = colBase + waveN * 64 + j * 16 + lm;
            const float bn = bias[col];
            #pragma unroll
            for (int r = 0; r < 8; ++r) {
                const int row = rowBase + waveM * 32 + i * 16 + hi * 8 + r;
                if (row < M) outf[(size_t)row * N + col] = acc[i][j][r] + bn;
            }
        }
    }
}

// ---------------------------------------------------------------------------
// Elementwise recurrences: one thread per (b,h), stream over t (coalesced).
// ---------------------------------------------------------------------------
__global__ __launch_bounds__(256) void scan1_kernel(
    half_t* __restrict__ P1, const half_t* __restrict__ C1)
{
    const int e = blockIdx.x * blockDim.x + threadIdx.x;
    float h = 0.0f;
    size_t off = e;
    #pragma unroll 4
    for (int t = 0; t < T_STEPS; ++t, off += (size_t)BATCH * HSD) {
        h = h * (float)C1[off] + (float)P1[off];
        P1[off] = (half_t)fast_tanh(h);
    }
}

__global__ __launch_bounds__(256) void scan2_kernel(
    const half_t* __restrict__ U2, const half_t* __restrict__ C2,
    half_t* __restrict__ H2T)
{
    const int e = blockIdx.x * blockDim.x + threadIdx.x;
    float h = 0.0f;
    size_t off = e;
    #pragma unroll 4
    for (int t = 0; t < T_STEPS; ++t, off += (size_t)BATCH * HSD) {
        h = h * (float)C2[off] + (float)U2[off];
    }
    H2T[e] = (half_t)fast_tanh(h);
}

// ---------------------------------------------------------------------------
extern "C" void kernel_launch(void* const* d_in, const int* in_sizes, int n_in,
                              void* d_out, int out_size, void* d_ws, size_t ws_size,
                              hipStream_t stream) {
    const float* data = (const float*)d_in[0];
    const float* W1   = (const float*)d_in[1];
    const float* b1   = (const float*)d_in[2];
    const float* Wi1  = (const float*)d_in[3];
    const float* bi1  = (const float*)d_in[4];
    const float* t1   = (const float*)d_in[5];
    const float* dr1  = (const float*)d_in[6];
    const float* W2   = (const float*)d_in[7];
    const float* b2   = (const float*)d_in[8];
    const float* Wi2  = (const float*)d_in[9];
    const float* bi2  = (const float*)d_in[10];
    const float* t2   = (const float*)d_in[11];
    const float* dr2  = (const float*)d_in[12];
    const float* W3   = (const float*)d_in[13];
    const float* b3   = (const float*)d_in[14];

    char*  ws  = (char*)d_ws;
    size_t off = 0;
    auto alloc = [&](size_t bytes) -> void* {
        void* p = ws + off;
        off += (bytes + 255) & ~(size_t)255;
        return p;
    };

    half_t* Xh   = (half_t*)alloc((size_t)MROWS * INPD * 2);
    half_t* W1h  = (half_t*)alloc((size_t)HSD * INPD * 2);
    half_t* Wi1h = (half_t*)alloc((size_t)HSD * INPD * 2);
    half_t* W2h  = (half_t*)alloc((size_t)HSD * HSD * 2);
    half_t* Wi2h = (half_t*)alloc((size_t)HSD * HSD * 2);
    half_t* W3h  = (half_t*)alloc((size_t)OUTD * HSD * 2);
    half_t* A1h  = (half_t*)alloc((size_t)MROWS * HSD * 2);
    half_t* C1h  = (half_t*)alloc((size_t)MROWS * HSD * 2);
    half_t* P1h  = (half_t*)alloc((size_t)MROWS * HSD * 2);   // -> an1 in place
    half_t* C2h  = (half_t*)alloc((size_t)MROWS * HSD * 2);
    half_t* U2h  = (half_t*)alloc((size_t)MROWS * HSD * 2);
    half_t* H2T  = (half_t*)alloc((size_t)BATCH * HSD * 2);

    cvt_f32_f16<<<4096, 256, 0, stream>>>(data, Xh,  MROWS * INPD);
    cvt_f32_f16<<<512,  256, 0, stream>>>(W1,   W1h,  HSD * INPD);
    cvt_f32_f16<<<512,  256, 0, stream>>>(Wi1,  Wi1h, HSD * INPD);
    cvt_f32_f16<<<512,  256, 0, stream>>>(W2,   W2h,  HSD * HSD);
    cvt_f32_f16<<<512,  256, 0, stream>>>(Wi2,  Wi2h, HSD * HSD);
    cvt_f32_f16<<<512,  256, 0, stream>>>(W3,   W3h,  OUTD * HSD);

    const dim3 gBig(HSD / 128, MROWS / 128);

    gemm_wmma_async<EPI_SIG_GATE><<<gBig, 256, 0, stream>>>(
        Xh, Wi1h, MROWS, HSD, INPD, bi1, t1, dr1, A1h, C1h);

    gemm_wmma_async<EPI_BIAS_F16><<<gBig, 256, 0, stream>>>(
        Xh, W1h, MROWS, HSD, INPD, b1, nullptr, nullptr, P1h, nullptr);

    gemm_wmma_async<EPI_GATE><<<gBig, 256, 0, stream>>>(
        A1h, Wi2h, MROWS, HSD, HSD, bi2, t2, dr2, nullptr, C2h);

    scan1_kernel<<<(BATCH * HSD) / 256, 256, 0, stream>>>(P1h, C1h);

    gemm_wmma_async<EPI_BIAS_F16><<<gBig, 256, 0, stream>>>(
        P1h, W2h, MROWS, HSD, HSD, b2, nullptr, nullptr, U2h, nullptr);

    scan2_kernel<<<(BATCH * HSD) / 256, 256, 0, stream>>>(U2h, C2h, H2T);

    gemm_wmma_small<<<dim3(OUTD / 128, 1), 256, 0, stream>>>(
        H2T, W3h, BATCH, OUTD, HSD, b3, (float*)d_out);
}
